// RelPositionMultiheadAttentionWeights_18124761989824
// MI455X (gfx1250) — matrix operand order repaired and checked by reading, not compiled, not run on hardware
//
#include <hip/hip_runtime.h>
#include <hip/hip_bf16.h>

typedef __attribute__((ext_vector_type(16))) _Float16 v16h;
typedef __attribute__((ext_vector_type(8)))  float    v8f;
typedef __attribute__((ext_vector_type(4)))  float    v4f;

#define Tq 512
#define Bq 64
#define Dq 512
#define Hq 5
#define QHDq 32
#define PHDq 4
#define NPOS 1023   // 2*T-1

// ---------------- workspace layout (bytes) ----------------
#define OFF_Q     0u                     // (B,H,T,32) f16  = 10485760
#define OFF_K     10485760u              // (B,H,S,32) f16  = 10485760
#define OFF_P     20971520u              // (B,H,T,4)  f32  = 2621440
#define OFF_PE    23592960u              // (H,1023,4) f32  = 81840 (pad to 81920)
#define OFF_WTLM  23674880u              // (192,512)  f16  = 196608
#define OFF_WTAM  23871488u              // (160,512)  f16  = 163840

// ---- CDNA5 async global->LDS DMA (ASYNCcnt-tracked) ----
__device__ __forceinline__ void async_ld_b128(unsigned lds_dst, const void* gsrc) {
    asm volatile("global_load_async_to_lds_b128 %0, %1, off"
                 :: "v"(lds_dst), "v"(gsrc) : "memory");
}
__device__ __forceinline__ void wait_asynccnt0() {
    asm volatile("s_wait_asynccnt 0x0" ::: "memory");
}
__device__ __forceinline__ unsigned lds_off(const void* p) {
    // low 32 bits of the flat address of a __shared__ object = LDS byte offset
    return (unsigned)(uintptr_t)p;
}

// ---------------------------------------------------------------------------
// Kernel 1: transpose + f16-convert weights, zero-pad lm to 192 cols
// ---------------------------------------------------------------------------
__global__ __launch_bounds__(256)
void prep_wt_kernel(const float* __restrict__ W_lm, const float* __restrict__ W_am,
                    _Float16* __restrict__ Wt_lm, _Float16* __restrict__ Wt_am)
{
    int i = blockIdx.x * 256 + threadIdx.x;
    if (i < 192 * 512) {
        int n = i >> 9, k = i & 511;
        float v = (n < 180) ? W_lm[(size_t)k * 180 + n] : 0.0f;
        Wt_lm[i] = (_Float16)v;
    } else {
        int j = i - 192 * 512;
        if (j < 160 * 512) {
            int n = j >> 9, k = j & 511;
            Wt_am[j] = (_Float16)W_am[(size_t)k * 160 + n];
        }
    }
}

// ---------------------------------------------------------------------------
// Kernel 2: pe[h][n][d] = sum_c pos_emb[n][c] * W_pos[c][h*4+d]
// ---------------------------------------------------------------------------
__global__ __launch_bounds__(256)
void pe_kernel(const float* __restrict__ pos_emb,  // (1023,192)
               const float* __restrict__ W_pos,    // (192,20)
               float* __restrict__ peout)          // (H,1023,4)
{
    int i = blockIdx.x * 256 + threadIdx.x;
    if (i >= NPOS * Hq * PHDq) return;
    int n = i / 20, c = i % 20;
    int h = c >> 2, d = c & 3;
    const float* pr = pos_emb + (size_t)n * 192;
    float acc = 0.0f;
    for (int kk = 0; kk < 192; ++kk) acc += pr[kk] * W_pos[kk * 20 + c];
    peout[((size_t)h * NPOS + n) * 4 + d] = acc;
}

// ---------------------------------------------------------------------------
// Kernel 3: projection GEMM (M=T*B, K=512, N<=Ntiles*16) via WMMA f16
// ---------------------------------------------------------------------------
__global__ __launch_bounds__(128)
void proj_kernel(const float* __restrict__ A,       // (32768, 512)
                 const _Float16* __restrict__ Wt,   // (Ntiles*16, 512)
                 const float* __restrict__ bias,    // (Ncols)
                 _Float16* __restrict__ qout,       // (B,H,T,32) f16 (or k)
                 float* __restrict__ pout,          // (B,H,T,4) f32 or null
                 int Ncols, int NtilesTotal)
{
    const int lane = threadIdx.x & 31;
    const int wave = threadIdx.x >> 5;
    const int n_tile = blockIdx.y * 4 + wave;
    if (n_tile >= NtilesTotal) return;          // wave-uniform exit

    const int m0   = blockIdx.x * 16;
    const int L    = lane & 15;                 // A row / B col within tile
    const int Mb   = (lane & 16) ? 8 : 0;       // C row base
    const int kb8  = (lane & 16) ? 8 : 0;       // unused (A frag K base is kb16)
    const int kb16 = (lane & 16) ? 16 : 0;      // B frag K base
    const int n0   = n_tile * 16;

    const float*    arow = A  + (size_t)(m0 + L) * Dq;
    const _Float16* brow = Wt + (size_t)(n0 + L) * Dq;

    v8f c = {0.f, 0.f, 0.f, 0.f, 0.f, 0.f, 0.f, 0.f};
    for (int k0 = 0; k0 < Dq; k0 += 32) {
        union { v16h v; _Float16 e[16]; } a;
        const float* pa = arow + k0;
        #pragma unroll
        for (int i = 0; i < 8; ++i) {
            a.e[i]     = (_Float16)pa[kb16 + i];       // K = kb16..kb16+7
            a.e[8 + i] = (_Float16)pa[kb16 + 8 + i];   // K = kb16+8..+15
        }
        v16h b = *(const v16h*)(brow + k0 + kb16);     // K = kb16..kb16+15
        c = __builtin_amdgcn_wmma_f32_16x16x32_f16(false, a.v, false, b,
                                                   (short)0, c, false, false);
    }

    const int n = n0 + L;
    const float bv = (n < Ncols) ? bias[n] : 0.0f;
    #pragma unroll
    for (int r = 0; r < 8; ++r) {
        int m  = m0 + Mb + r;
        int t  = m >> 6;            // / B
        int bb = m & 63;            // % B
        float val = c[r] + bv;
        if (n < 160) {
            int h = n >> 5, d = n & 31;
            qout[((((size_t)bb * Hq + h) * Tq) + t) * 32 + d] = (_Float16)val;
        } else if (n < Ncols && pout) {
            int h = (n - 160) >> 2, d = (n - 160) & 3;
            pout[((((size_t)bb * Hq + h) * Tq) + t) * 4 + d] = val;
        }
    }
}

// ---------------------------------------------------------------------------
// Kernel 4: attention scores + rel-shift pos term + mask + softmax
//   one block = (h, b, 16 query rows); 8 waves x 4 s-tiles of 16.
//   pe/p/mask staged into LDS via async DMA, overlapped with WMMA.
// ---------------------------------------------------------------------------
#define LDSS 516
__global__ __launch_bounds__(256)
void attn_kernel(const _Float16* __restrict__ qbuf,  // (B,H,T,32)
                 const _Float16* __restrict__ kbuf,  // (B,H,S,32)
                 const float* __restrict__ pbuf,     // (B,H,T,4)
                 const float* __restrict__ pebuf,    // (H,1023,4)
                 const unsigned char* __restrict__ mask, // (B,S) bool
                 float* __restrict__ out)            // (H,B,T,S)
{
    __shared__ __align__(16) float pe_sh[NPOS * 4];
    __shared__ __align__(16) float p_sh[16 * 4];
    __shared__ __align__(16) float sc[16 * LDSS];
    __shared__ __align__(16) unsigned char m_sh[512];
    __shared__ float red[16 * 16];
    __shared__ float rowv[16];

    const int tid  = threadIdx.x;
    const int lane = tid & 31;
    const int wave = tid >> 5;
    const int bx = blockIdx.x;
    const int t0 = (bx & 31) * 16;
    const int b  = (bx >> 5) & 63;
    const int h  = bx >> 11;

    // ---- launch async DMA: pe table (16368 B), p rows (256 B), mask (512 B)
    {
        const float* peh = pebuf + (size_t)h * NPOS * 4;
        const unsigned pe_base = lds_off(pe_sh);
        for (int ch = tid; ch < NPOS; ch += 256)            // 1023 x 16B
            async_ld_b128(pe_base + ch * 16, peh + ch * 4);
        if (tid < 16) {                                     // 16 x 16B
            const float* pr = pbuf + (((size_t)b * Hq + h) * Tq + t0) * 4;
            async_ld_b128(lds_off(p_sh) + tid * 16, pr + tid * 4);
        }
        if (tid < 32) {                                     // 32 x 16B
            const unsigned char* mr = mask + (size_t)b * Tq;
            async_ld_b128(lds_off(m_sh) + tid * 16, mr + tid * 16);
        }
    }

    // ---- WMMA phase (overlaps the DMA; touches only global + VGPRs)
    const _Float16* qb    = qbuf + (((size_t)b * Hq + h) * Tq + t0) * 32;
    const _Float16* kbase = kbuf + ((size_t)b * Hq + h) * Tq * 32;
    const int L  = lane & 15;
    const int Mb = (lane & 16) ? 8 : 0;

    union { v16h v; _Float16 e[16]; } a;
    {
        const _Float16* ap = qb + (size_t)L * 32 + ((lane & 16) ? 16 : 0);
        #pragma unroll
        for (int i = 0; i < 8; ++i) { a.e[i] = ap[i]; a.e[8 + i] = ap[8 + i]; }
    }
    v16h bf[4];
    #pragma unroll
    for (int j = 0; j < 4; ++j) {
        const int s0 = (wave * 4 + j) * 16;
        bf[j] = *(const v16h*)(kbase + (size_t)(s0 + L) * 32 + ((lane & 16) ? 16 : 0));
    }
    v8f cacc[4];
    #pragma unroll
    for (int j = 0; j < 4; ++j) {
        v8f z = {0.f, 0.f, 0.f, 0.f, 0.f, 0.f, 0.f, 0.f};
        cacc[j] = __builtin_amdgcn_wmma_f32_16x16x32_f16(false, a.v, false, bf[j],
                                                         (short)0, z, false, false);
    }

    // ---- DMA must be complete before reading pe/p/mask from LDS
    wait_asynccnt0();
    __syncthreads();

    #pragma unroll
    for (int j = 0; j < 4; ++j) {
        const int s = (wave * 4 + j) * 16 + L;
        #pragma unroll
        for (int r = 0; r < 8; ++r) {
            int row = Mb + r;
            int t   = t0 + row;
            int idx = s - t + (Tq - 1);          // always in [0,1022]
            v4f pe4 = *(const v4f*)&pe_sh[idx * 4];
            v4f pv4 = *(const v4f*)&p_sh[row * 4];
            float pos = pv4[0]*pe4[0] + pv4[1]*pe4[1] + pv4[2]*pe4[2] + pv4[3]*pe4[3];
            float val = cacc[j][r] + pos;
            if (m_sh[s]) val = -1000.0f;
            sc[row * LDSS + s] = val;
        }
    }
    __syncthreads();

    // ---- softmax over s (512) per row ----
    const int row = tid >> 4;
    const int seg = tid & 15;
    float* srow = &sc[row * LDSS + seg * 32];
    float mx = -3.0e38f;
    #pragma unroll 8
    for (int i = 0; i < 32; ++i) mx = fmaxf(mx, srow[i]);
    red[row * 16 + seg] = mx;
    __syncthreads();
    if (tid < 16) {
        float m2 = -3.0e38f;
        #pragma unroll
        for (int i = 0; i < 16; ++i) m2 = fmaxf(m2, red[tid * 16 + i]);
        rowv[tid] = m2;
    }
    __syncthreads();
    const float rmax = rowv[row];
    float sum = 0.0f;
    #pragma unroll 8
    for (int i = 0; i < 32; ++i) {
        float e = __expf(srow[i] - rmax);
        srow[i] = e;
        sum += e;
    }
    red[row * 16 + seg] = sum;
    __syncthreads();
    if (tid < 16) {
        float s2 = 0.0f;
        #pragma unroll
        for (int i = 0; i < 16; ++i) s2 += red[tid * 16 + i];
        rowv[tid] = 1.0f / s2;
    }
    __syncthreads();

    // ---- coalesced, non-temporal output stream (written once, never reread)
    float* ob = out + (((size_t)h * Bq + b) * Tq + t0) * (size_t)Tq;
    #pragma unroll 8
    for (int i = 0; i < 32; ++i) {
        int idx = i * 256 + tid;
        int r2 = idx >> 9, c2 = idx & 511;
        __builtin_nontemporal_store(sc[r2 * LDSS + c2] * rowv[r2],
                                    &ob[(size_t)r2 * Tq + c2]);
    }
}

// ---------------------------------------------------------------------------
extern "C" void kernel_launch(void* const* d_in, const int* in_sizes, int n_in,
                              void* d_out, int out_size, void* d_ws, size_t ws_size,
                              hipStream_t stream)
{
    const float* lm       = (const float*)d_in[0];
    const float* am       = (const float*)d_in[1];
    const float* pos_emb  = (const float*)d_in[2];
    const unsigned char* mask = (const unsigned char*)d_in[3];
    const float* W_lm     = (const float*)d_in[4];
    const float* b_lm     = (const float*)d_in[5];
    const float* W_am     = (const float*)d_in[6];
    const float* b_am     = (const float*)d_in[7];
    const float* W_pos    = (const float*)d_in[8];
    float* out = (float*)d_out;

    char* ws = (char*)d_ws;
    _Float16* qbuf  = (_Float16*)(ws + OFF_Q);
    _Float16* kbuf  = (_Float16*)(ws + OFF_K);
    float*    pbuf  = (float*)(ws + OFF_P);
    float*    pebuf = (float*)(ws + OFF_PE);
    _Float16* WtLm  = (_Float16*)(ws + OFF_WTLM);
    _Float16* WtAm  = (_Float16*)(ws + OFF_WTAM);

    // 1. weight transpose/convert: (192+160)*512 = 180224 elems
    prep_wt_kernel<<<704, 256, 0, stream>>>(W_lm, W_am, WtLm, WtAm);
    // 2. positional embedding projection: 1023*20 = 20460 elems
    pe_kernel<<<80, 256, 0, stream>>>(pos_emb, W_pos, pebuf);
    // 3. lm projection: M tiles = 32768/16 = 2048, N tiles = 12 (4 waves/blk)
    proj_kernel<<<dim3(2048, 3), 128, 0, stream>>>(lm, WtLm, b_lm, qbuf, pbuf, 180, 12);
    // 4. am projection: N tiles = 10
    proj_kernel<<<dim3(2048, 3), 128, 0, stream>>>(am, WtAm, b_am, kbuf, nullptr, 160, 10);
    // 5. attention: H*B*(T/16) = 5*64*32 = 10240 blocks
    attn_kernel<<<10240, 256, 0, stream>>>(qbuf, kbuf, pbuf, pebuf, mask, out);
}
